// SelfAttention_47098611368442
// MI455X (gfx1250) — matrix operand order repaired and checked by reading, not compiled, not run on hardware
//
#include <hip/hip_runtime.h>

// ---------------------------------------------------------------------------
// MI455X / gfx1250 self-attention (GQA + KV cache).
// f32->bf16 pre-passes (bandwidth-cheap at 23.3 TB/s), then pure bf16 WMMA
// with packed b128 fragment loads; V fragments software-pipelined ahead of
// softmax; 32x64 GEMM wave tiles for 2x operand reuse.
// ---------------------------------------------------------------------------

typedef __attribute__((ext_vector_type(16))) __bf16 v16bf;
typedef __attribute__((ext_vector_type(16))) unsigned short v16u;
typedef __attribute__((ext_vector_type(8))) unsigned short v8u;
typedef __attribute__((ext_vector_type(4))) unsigned short v4u;
typedef __attribute__((ext_vector_type(8))) float v8f;

union FragBF {
  v16u us;
  v16bf bf;
  struct { v8u lo, hi; } h;
};

__device__ __forceinline__ unsigned short f2bf(float f) {
  unsigned u = __builtin_bit_cast(unsigned, f);
  u += 0x7FFFu + ((u >> 16) & 1u);  // round-to-nearest-even
  return (unsigned short)(u >> 16);
}
__device__ __forceinline__ float bf2f(unsigned short u) {
  return __builtin_bit_cast(float, (unsigned)u << 16);
}

constexpr int kDim = 4096;
constexpr int kHeads = 32;
constexpr int kKvHeads = 8;
constexpr int kHeadDim = 128;
constexpr int kBatch = 4;
constexpr int kSeq = 512;
constexpr int kMaxSeq = 4096;
constexpr int kStart = 2048;                 // START_POS (multiple of 32)
constexpr int kTot = kStart + kSeq;          // 2560
constexpr int kRows = kBatch * kSeq;         // 2048
constexpr int kKvDim = kKvHeads * kHeadDim;  // 1024

// ---------------------------------------------------------------------------
// f32 -> packed bf16, 4 elements/thread.
// ---------------------------------------------------------------------------
__global__ void cvt_f32_bf16(const float* __restrict__ src,
                             unsigned short* __restrict__ dst, int n4) {
  int i = blockIdx.x * blockDim.x + threadIdx.x;
  if (i >= n4) return;
  float4 v = ((const float4*)src)[i];
  v4u o;
  o.x = f2bf(v.x); o.y = f2bf(v.y); o.z = f2bf(v.z); o.w = f2bf(v.w);
  ((v4u*)dst)[i] = o;
}

// ---------------------------------------------------------------------------
// Build transposed bf16 V: vtb[b][kvh][d][t] for t in [0,kTot).
// t <  kStart : from f32 input cache_v[b][t][kvh][d]
// t >= kStart : from bf16 fresh V rows vbn[(b*kSeq+s)*kKvDim + kvh*128 + d]
// Output-linear indexing -> coalesced 2B stores; reads ride the 192MB L2.
// ---------------------------------------------------------------------------
__global__ void build_vT(const float* __restrict__ cache_v,
                         const unsigned short* __restrict__ vbn,
                         unsigned short* __restrict__ vtb, int total) {
  int idx = blockIdx.x * blockDim.x + threadIdx.x;
  if (idx >= total) return;
  int t = idx % kTot;
  int tmp = idx / kTot;
  int d = tmp % kHeadDim;
  tmp /= kHeadDim;
  int kvh = tmp % kKvHeads;
  int b = tmp / kKvHeads;
  unsigned short us;
  if (t < kStart) {
    us = f2bf(cache_v[(((size_t)b * kMaxSeq + t) * kKvHeads + kvh) * kHeadDim + d]);
  } else {
    us = vbn[((size_t)(b * kSeq + (t - kStart)) * kKvHeads + kvh) * kHeadDim + d];
  }
  vtb[idx] = us;
}

// ---------------------------------------------------------------------------
// GEMM: Out[M,N] = A[M,K] (bf16) * W[N,K]^T (bf16), f32 WMMA accumulate.
// One wave owns a 32x64 tile: 2 A fragments x 4 B fragments -> 8 WMMAs per
// K=32 chunk with each loaded fragment reused (2x the FLOP/byte of 16x64).
// Fragment layouts as loaded here (wave32; contiguous k, same order for A and B):
//   A 16x32: lane%16 = M row; lane half selects K 0..15 / 16..31.
//   B 32x16: lane%16 = N col; lane half selects K 0..15 / 16..31.
//   C/D    : vgpr r -> M = r + 8*half, N = lane%16.
// ---------------------------------------------------------------------------
template <bool OUT_F32>
__global__ __launch_bounds__(32)
void gemm_bf16_wmma(const unsigned short* __restrict__ A,
                    const unsigned short* __restrict__ W,
                    void* __restrict__ Outv, int N, int K) {
  const int lane = threadIdx.x & 31;
  const int col = lane & 15;
  const int half = lane >> 4;
  const int m0 = blockIdx.y * 32;
  const int n0 = blockIdx.x * 64;

  v8f zero = {};
  v8f acc[2][4];
#pragma unroll
  for (int mt = 0; mt < 2; ++mt)
#pragma unroll
    for (int nt = 0; nt < 4; ++nt) acc[mt][nt] = zero;

  const unsigned short* arow0 = A + (size_t)(m0 + col) * K;
  const unsigned short* arow1 = A + (size_t)(m0 + 16 + col) * K;
  for (int k0 = 0; k0 < K; k0 += 32) {
    FragBF a[2];
    a[0].h.lo = *(const v8u*)(arow0 + k0 + half * 16);
    a[0].h.hi = *(const v8u*)(arow0 + k0 + half * 16 + 8);
    a[1].h.lo = *(const v8u*)(arow1 + k0 + half * 16);
    a[1].h.hi = *(const v8u*)(arow1 + k0 + half * 16 + 8);
#pragma unroll
    for (int nt = 0; nt < 4; ++nt) {
      FragBF b;
      b.us = *(const v16u*)(W + (size_t)(n0 + nt * 16 + col) * K + k0 + half * 16);
#pragma unroll
      for (int mt = 0; mt < 2; ++mt) {
        acc[mt][nt] = __builtin_amdgcn_wmma_f32_16x16x32_bf16(
            false, a[mt].bf, false, b.bf, (short)0, acc[mt][nt], false, false);
      }
    }
  }

#pragma unroll
  for (int mt = 0; mt < 2; ++mt) {
#pragma unroll
    for (int nt = 0; nt < 4; ++nt) {
#pragma unroll
      for (int r = 0; r < 8; ++r) {
        size_t o = (size_t)(m0 + mt * 16 + half * 8 + r) * N + n0 + nt * 16 + col;
        if constexpr (OUT_F32) ((float*)Outv)[o] = acc[mt][nt][r];
        else ((unsigned short*)Outv)[o] = f2bf(acc[mt][nt][r]);
      }
    }
  }
}

// ---------------------------------------------------------------------------
// In-place RoPE on bf16 [kRows, heads, 128] (interleaved pairs).
// ---------------------------------------------------------------------------
__global__ void rope_bf16(unsigned short* __restrict__ t,
                          const float* __restrict__ cosT,
                          const float* __restrict__ sinT, int heads, int total) {
  int idx = blockIdx.x * blockDim.x + threadIdx.x;
  if (idx >= total) return;
  int j = idx & 63;
  int tmp = idx >> 6;
  int h = tmp % heads;
  int row = tmp / heads;  // b*kSeq + s
  int s = row % kSeq;
  float c = cosT[s * 64 + j];
  float sn = sinT[s * 64 + j];
  unsigned short* p = t + ((size_t)row * heads + h) * kHeadDim + 2 * j;
  float xr = bf2f(p[0]), xi = bf2f(p[1]);
  p[0] = f2bf(xr * c - xi * sn);
  p[1] = f2bf(xr * sn + xi * c);
}

// ---------------------------------------------------------------------------
// Flash attention, one wave per (batch, head, 16-query tile).
// Per 32-wide T chunk:
//   1) issue all 8 V fragment loads (independent of softmax)
//   2) QK^T = 8 wmma  -> LDS online softmax (exp hides V load latency)
//   3) PV = 8 wmma on the preloaded V fragments
// K rows: bf16, row stride kKvDim (cached region kcb / fresh region kbn).
// V: transposed bf16 vtb[b][kvh][d][t] -> contiguous-t b128 fragment loads.
// Writes bf16 attention output (input of the wo projection).
// ---------------------------------------------------------------------------
__global__ __launch_bounds__(32)
void attn_wmma(const unsigned short* __restrict__ qb,
               const unsigned short* __restrict__ kcb,
               const unsigned short* __restrict__ kbn,
               const unsigned short* __restrict__ vtb,
               unsigned short* __restrict__ out) {
  __shared__ float s_scores[16][32];
  __shared__ unsigned short s_p[16][32];
  __shared__ float s_corr[16];
  __shared__ float s_sum[16];

  const int lane = threadIdx.x & 31;
  const int col = lane & 15;
  const int half = lane >> 4;
  const int q0 = blockIdx.x * 16;
  const int h = blockIdx.y;
  const int bb = blockIdx.z;
  const int kvh = h >> 2;  // N_REP = 4

  const float scale = 0.08838834764831845f;  // 1/sqrt(128)
  v8f zero = {};

  // Q A-fragments: 16 rows x 128 K as 4 chunks of 32 (pure b128 loads).
  FragBF qa[4];
  const unsigned short* qrow =
      qb + (size_t)(bb * kSeq + q0 + col) * kDim + h * kHeadDim;
#pragma unroll
  for (int kc = 0; kc < 4; ++kc) {
    qa[kc].h.lo = *(const v8u*)(qrow + kc * 32 + half * 16);
    qa[kc].h.hi = *(const v8u*)(qrow + kc * 32 + half * 16 + 8);
  }

  const unsigned short* vhead = vtb + ((size_t)(bb * kKvHeads + kvh) * kHeadDim) * kTot;

  v8f oacc[8];
#pragma unroll
  for (int dt = 0; dt < 8; ++dt) oacc[dt] = zero;

  float m_prev = -3.0e38f;
  float sum = 0.f;

  for (int t0 = 0; t0 < kTot; t0 += 32) {
    // ---- stage V fragments early: latency hidden by QK^T + softmax ----
    FragBF vb[8];
#pragma unroll
    for (int dt = 0; dt < 8; ++dt) {
      // B[k][n] = V[t0+k][dt*16+n] = vT[dt*16+n][t0+k], contiguous in t
      vb[dt].us = *(const v16u*)(vhead + (size_t)(dt * 16 + col) * kTot + t0 + half * 16);
    }

    const unsigned short* kbase =
        (t0 < kStart)
            ? kcb + (((size_t)bb * kStart + t0) * kKvHeads + kvh) * kHeadDim
            : kbn + ((size_t)(bb * kSeq + (t0 - kStart)) * kKvHeads + kvh) * kHeadDim;

    // scores(16x32) = Q(16x128) * K_chunk^T(128x32)
#pragma unroll
    for (int st = 0; st < 2; ++st) {
      v8f sacc = zero;
      const unsigned short* krow = kbase + (size_t)(st * 16 + col) * kKvDim;
#pragma unroll
      for (int kc = 0; kc < 4; ++kc) {
        FragBF kb;
        kb.us = *(const v16u*)(krow + kc * 32 + half * 16);
        sacc = __builtin_amdgcn_wmma_f32_16x16x32_bf16(
            false, qa[kc].bf, false, kb.bf, (short)0, sacc, false, false);
      }
#pragma unroll
      for (int r = 0; r < 8; ++r)
        s_scores[half * 8 + r][st * 16 + col] = sacc[r] * scale;
    }
    __syncthreads();

    // Online softmax: lanes 0..15 each own one query row.
    if (lane < 16) {
      float mx = m_prev;
#pragma unroll 8
      for (int j = 0; j < 32; ++j) mx = fmaxf(mx, s_scores[lane][j]);
      float corr = __expf(m_prev - mx);
      float ls = 0.f;
#pragma unroll 8
      for (int j = 0; j < 32; ++j) {
        float p = __expf(s_scores[lane][j] - mx);
        s_p[lane][j] = f2bf(p);
        ls += p;
      }
      sum = sum * corr + ls;
      m_prev = mx;
      s_corr[lane] = corr;
    }
    __syncthreads();

    // O = O*corr + P(16x32) * V_chunk(32x128); V already in registers.
    FragBF pa;
    pa.h.lo = *(const v8u*)&s_p[col][half * 16];
    pa.h.hi = *(const v8u*)&s_p[col][half * 16 + 8];
#pragma unroll
    for (int dt = 0; dt < 8; ++dt) {
#pragma unroll
      for (int r = 0; r < 8; ++r) oacc[dt][r] *= s_corr[half * 8 + r];
      oacc[dt] = __builtin_amdgcn_wmma_f32_16x16x32_bf16(
          false, pa.bf, false, vb[dt].bf, (short)0, oacc[dt], false, false);
    }
    __syncthreads();  // protect s_p / s_corr before next chunk
  }

  if (lane < 16) s_sum[lane] = sum;
  __syncthreads();

#pragma unroll
  for (int dt = 0; dt < 8; ++dt) {
#pragma unroll
    for (int r = 0; r < 8; ++r) {
      int row = half * 8 + r;
      out[(size_t)(bb * kSeq + q0 + row) * kDim + h * kHeadDim + dt * 16 + col] =
          f2bf(oacc[dt][r] / s_sum[row]);
    }
  }
}

// ---------------------------------------------------------------------------
extern "C" void kernel_launch(void* const* d_in, const int* in_sizes, int n_in,
                              void* d_out, int out_size, void* d_ws, size_t ws_size,
                              hipStream_t stream) {
  (void)in_sizes; (void)n_in; (void)out_size; (void)ws_size;
  const float* x = (const float*)d_in[0];
  const float* cache_k = (const float*)d_in[1];
  const float* cache_v = (const float*)d_in[2];
  const float* wq = (const float*)d_in[3];
  const float* wk = (const float*)d_in[4];
  const float* wv = (const float*)d_in[5];
  const float* wo = (const float*)d_in[6];
  const float* fcos = (const float*)d_in[7];
  const float* fsin = (const float*)d_in[8];
  float* out = (float*)d_out;

  // Workspace layout (bf16 = unsigned short elements), ~173 MB total.
  unsigned short* ws = (unsigned short*)d_ws;
  unsigned short* xb  = ws;                               // 2048*4096
  unsigned short* wqb = xb  + (size_t)kRows * kDim;       // 4096*4096
  unsigned short* wkb = wqb + (size_t)kDim * kDim;        // 1024*4096
  unsigned short* wvb = wkb + (size_t)kKvDim * kDim;      // 1024*4096
  unsigned short* wob = wvb + (size_t)kKvDim * kDim;      // 4096*4096
  unsigned short* qbuf = wob + (size_t)kDim * kDim;       // 2048*4096
  unsigned short* kbn = qbuf + (size_t)kRows * kDim;      // 2048*1024
  unsigned short* vbn = kbn + (size_t)kRows * kKvDim;     // 2048*1024
  unsigned short* kcb = vbn + (size_t)kRows * kKvDim;     // 4*2048*1024
  unsigned short* vtb = kcb + (size_t)kBatch * kStart * kKvDim;  // 4*8*128*2560
  unsigned short* ab  = vtb + (size_t)kBatch * kKvDim * kTot;    // 2048*4096

  dim3 wave(32);
  auto cvt = [&](const float* s, unsigned short* d, size_t n) {
    int n4 = (int)(n / 4);
    cvt_f32_bf16<<<(n4 + 255) / 256, 256, 0, stream>>>(s, d, n4);
  };

  // --- Pre-pass: pack everything hot into bf16 ---
  cvt(x, xb, (size_t)kRows * kDim);
  cvt(wq, wqb, (size_t)kDim * kDim);
  cvt(wk, wkb, (size_t)kKvDim * kDim);
  cvt(wv, wvb, (size_t)kKvDim * kDim);
  cvt(wo, wob, (size_t)kDim * kDim);
  for (int b = 0; b < kBatch; ++b) {  // cached K region, per batch (t < kStart)
    cvt(cache_k + (size_t)b * kMaxSeq * kKvDim,
        kcb + (size_t)b * kStart * kKvDim, (size_t)kStart * kKvDim);
  }

  // --- QKV projections (bf16 WMMA, bf16 out) ---
  gemm_bf16_wmma<false><<<dim3(kDim / 64, kRows / 32), wave, 0, stream>>>(
      xb, wqb, qbuf, kDim, kDim);
  gemm_bf16_wmma<false><<<dim3(kKvDim / 64, kRows / 32), wave, 0, stream>>>(
      xb, wkb, kbn, kKvDim, kDim);
  gemm_bf16_wmma<false><<<dim3(kKvDim / 64, kRows / 32), wave, 0, stream>>>(
      xb, wvb, vbn, kKvDim, kDim);

  // --- RoPE (in-place, bf16) ---
  int totalq = kRows * kHeads * (kHeadDim / 2);
  rope_bf16<<<(totalq + 255) / 256, 256, 0, stream>>>(qbuf, fcos, fsin, kHeads, totalq);
  int totalk = kRows * kKvHeads * (kHeadDim / 2);
  rope_bf16<<<(totalk + 255) / 256, 256, 0, stream>>>(kbn, fcos, fsin, kKvHeads, totalk);

  // --- Transposed bf16 V over full KV length ---
  int vtotal = kBatch * kKvHeads * kHeadDim * kTot;
  build_vT<<<(vtotal + 255) / 256, 256, 0, stream>>>(cache_v, vbn, vtb, vtotal);

  // --- Flash attention ---
  attn_wmma<<<dim3(kSeq / 16, kHeads, kBatch), wave, 0, stream>>>(
      qbuf, kcb, kbn, vtb, ab);

  // --- Output projection -> f32 d_out ---
  gemm_bf16_wmma<true><<<dim3(kDim / 64, kRows / 32), wave, 0, stream>>>(
      ab, wob, out, kDim, kDim);
}
